// ARPrior_34488587387582
// MI455X (gfx1250) — matrix operand order repaired and checked by reading, not compiled, not run on hardware
//
#include <hip/hip_runtime.h>
#include <hip/hip_bf16.h>

// ---------------------------------------------------------------------------
// ARPrior on gfx1250: bf16 WMMA (f32 accumulate) pipeline, M=32 per wave.
//   L=31 latent steps, B=65536 batch. Per step:
//     h1 = relu(z[:, :31] @ W1m[l]^T + b1[l])    (B x 128)
//     h2 = relu(h1 @ W2[l]^T + b2[l])            (B x 64)
//     out = h2 @ Wout[l]^T + bout[l]             (B x 2)
// Each wave processes TWO 16-row M-tiles so every B-fragment load feeds two
// independent WMMAs (2x arithmetic intensity on L2, fills XDL hazard slots).
// Weights repacked once per launch into WMMA B-fragment order (bf16), causal
// mask folded into W1, Wout zero-padded N 2->16 so layer 3 also runs on WMMA.
// ---------------------------------------------------------------------------

#define BATCH   65536
#define LSTEPS  31

typedef __attribute__((ext_vector_type(16))) __bf16 v16bf;
typedef __attribute__((ext_vector_type(8)))  float  v8f;

union Frag {
    v16bf v;
    uint4 q[2];
    unsigned short s[16];
};

__device__ __forceinline__ unsigned short f2bf(float f) {
    unsigned u = __float_as_uint(f);
    u += 0x7FFFu + ((u >> 16) & 1u);   // round-to-nearest-even
    return (unsigned short)(u >> 16);
}

// Wave-local LDS fence: compiler ordering + wait for all DS ops to complete.
// Each wave only touches its own LDS tiles, so no workgroup barrier is needed.
__device__ __forceinline__ void lds_fence() {
    __builtin_amdgcn_wave_barrier();
    asm volatile("s_wait_dscnt 0" ::: "memory");
    __builtin_amdgcn_wave_barrier();
}

// --------------------------- weight repack kernels --------------------------
// B-fragment layout for v_wmma_f32_16x16x32_bf16 (wave32):
//   lane c (0..15)  holds column n=c,  K = 0..7, 16..23  (V0..V7, 2 bf16/VGPR)
//   lane 16+c       holds column n=c,  K = 8..15, 24..31
// Fragment stored as 32 contiguous bytes per lane: [lane][e] e=0..15 -> K.

__global__ void prep_w1(const float* __restrict__ W1, unsigned short* __restrict__ w1f) {
    int t = blockIdx.x * 256 + threadIdx.x;            // 31*8*32*16 = 126976
    if (t >= LSTEPS * 8 * 32 * 16) return;
    int e = t & 15, lane = (t >> 4) & 31, nt = (t >> 9) & 7, l = t >> 12;
    int n = nt * 16 + (lane & 15);                     // h index (0..127)
    int K = ((e >> 3) << 4) + ((lane >> 4) << 3) + (e & 7); // d index (0..31)
    float v = (K <= l) ? W1[(l * 128 + n) * 31 + K] : 0.f;  // mask d<=l; K=31 -> 0
    w1f[t] = f2bf(v);
}

__global__ void prep_w2(const float* __restrict__ W2, unsigned short* __restrict__ w2f) {
    int t = blockIdx.x * 256 + threadIdx.x;            // 31*4*4*32*16 = 253952
    if (t >= LSTEPS * 4 * 4 * 32 * 16) return;
    int e = t & 15, lane = (t >> 4) & 31, ks = (t >> 9) & 3, nt = (t >> 11) & 3, l = t >> 13;
    int n = nt * 16 + (lane & 15);                     // k_out index (0..63)
    int K = ks * 32 + ((e >> 3) << 4) + ((lane >> 4) << 3) + (e & 7); // h index (0..127)
    w2f[t] = f2bf(W2[(l * 64 + n) * 128 + K]);
}

__global__ void prep_wo(const float* __restrict__ Wout, unsigned short* __restrict__ wof) {
    int t = blockIdx.x * 256 + threadIdx.x;            // 31*2*32*16 = 31744
    if (t >= LSTEPS * 2 * 32 * 16) return;
    int e = t & 15, lane = (t >> 4) & 31, ks = (t >> 9) & 1, l = t >> 10;
    int n = lane & 15;                                 // o index padded to 16
    int K = ks * 32 + ((e >> 3) << 4) + ((lane >> 4) << 3) + (e & 7); // k index (0..63)
    float v = (n < 2) ? Wout[(l * 2 + n) * 64 + K] : 0.f;
    wof[t] = f2bf(v);
}

// -------------------------------- main kernel --------------------------------

__global__ __launch_bounds__(128) void arprior_main(
    const float* __restrict__ z,
    const float* __restrict__ b1, const float* __restrict__ b2,
    const float* __restrict__ bout, const float* __restrict__ bout0,
    const unsigned short* __restrict__ w1f,
    const unsigned short* __restrict__ w2f,
    const unsigned short* __restrict__ wof,
    float* __restrict__ out)
{
    // Per-wave private activation tiles: 2 x (16x128 bf16); h2 reuses space.
    __shared__ __align__(16) unsigned short hs[4][2][16 * 128];

    const int lane = threadIdx.x & 31;
    const int wv   = threadIdx.x >> 5;          // 0..3
    const int tile = blockIdx.x * 4 + wv;       // 32-row macro tile
    const int rowBase = tile * 32;
    const int col  = lane & 15;                 // A row (M) / B,C,D column (N)
    const int half = lane >> 4;
    unsigned short* hA = hs[wv][0];
    unsigned short* hB = hs[wv][1];

    // Column 0 of mus/lvs = broadcast bout0 (32 rows -> full wave).
    {
        int r = rowBase + lane;
        out[r * 32]              = bout0[0];
        out[BATCH * 32 + r * 32] = bout0[1];
    }

    // z A-fragments (loop invariant). 16-bit A layout:
    //   lanes 0-15 : M=lane, K in {0..7, 16..23}
    //   lanes 16-31: M=lane-16, K in {8..15, 24..31}
    Frag aZ0, aZ1;
    {
        const float* zp0 = z + (size_t)(rowBase + col) * 32;
        const float* zp1 = zp0 + 16 * 32;
        #pragma unroll
        for (int e = 0; e < 8; ++e) {
            aZ0.s[e]     = f2bf(zp0[half * 8 + e]);
            aZ0.s[8 + e] = f2bf(zp0[16 + half * 8 + e]);
            aZ1.s[e]     = f2bf(zp1[half * 8 + e]);
            aZ1.s[8 + e] = f2bf(zp1[16 + half * 8 + e]);
        }
    }

    for (int l = 0; l < LSTEPS; ++l) {
        // ---- Layer 1: (16x32) x (32x128), 8 N-tiles, 2 M-tiles -----------
        #pragma unroll
        for (int nt = 0; nt < 8; ++nt) {
            Frag bW;
            const uint4* bp = (const uint4*)(w1f + (((l * 8 + nt) * 32 + lane) << 4));
            bW.q[0] = bp[0]; bW.q[1] = bp[1];
            v8f acc0 = {}, acc1 = {};
            acc0 = __builtin_amdgcn_wmma_f32_16x16x32_bf16(
                false, aZ0.v, false, bW.v, (short)0, acc0, false, false);
            acc1 = __builtin_amdgcn_wmma_f32_16x16x32_bf16(
                false, aZ1.v, false, bW.v, (short)0, acc1, false, false);
            float bias = b1[l * 128 + nt * 16 + col];
            #pragma unroll
            for (int j = 0; j < 8; ++j) {
                int m = j + half * 8;                       // D row
                float h0 = acc0[j] + bias; h0 = h0 > 0.f ? h0 : 0.f;
                float h1 = acc1[j] + bias; h1 = h1 > 0.f ? h1 : 0.f;
                hA[m * 128 + nt * 16 + col] = f2bf(h0);     // h1 tile 0
                hB[m * 128 + nt * 16 + col] = f2bf(h1);     // h1 tile 1
            }
        }
        lds_fence();

        // ---- h1 A-fragments (transpose via LDS), both M-tiles ------------
        Frag aH0[4], aH1[4];
        #pragma unroll
        for (int ks = 0; ks < 4; ++ks) {
            int o0 = col * 128 + ks * 32 + half * 8;
            aH0[ks].q[0] = *(const uint4*)&hA[o0];
            aH0[ks].q[1] = *(const uint4*)&hA[o0 + 16];
            aH1[ks].q[0] = *(const uint4*)&hB[o0];
            aH1[ks].q[1] = *(const uint4*)&hB[o0 + 16];
        }

        // ---- Layer 2: (16x128) x (128x64), K in 4 steps, 4 N-tiles -------
        #pragma unroll
        for (int nt = 0; nt < 4; ++nt) {
            v8f acc0 = {}, acc1 = {};
            #pragma unroll
            for (int ks = 0; ks < 4; ++ks) {
                Frag bW;
                const uint4* bp = (const uint4*)(w2f + ((((l * 4 + nt) * 4 + ks) * 32 + lane) << 4));
                bW.q[0] = bp[0]; bW.q[1] = bp[1];
                acc0 = __builtin_amdgcn_wmma_f32_16x16x32_bf16(
                    false, aH0[ks].v, false, bW.v, (short)0, acc0, false, false);
                acc1 = __builtin_amdgcn_wmma_f32_16x16x32_bf16(
                    false, aH1[ks].v, false, bW.v, (short)0, acc1, false, false);
            }
            float bias = b2[l * 64 + nt * 16 + col];
            #pragma unroll
            for (int j = 0; j < 8; ++j) {
                int m = j + half * 8;
                float h0 = acc0[j] + bias; h0 = h0 > 0.f ? h0 : 0.f;
                float h1 = acc1[j] + bias; h1 = h1 > 0.f ? h1 : 0.f;
                hA[m * 64 + nt * 16 + col] = f2bf(h0);      // h2 tile 0 (reuse LDS)
                hB[m * 64 + nt * 16 + col] = f2bf(h1);      // h2 tile 1
            }
        }
        lds_fence();

        // ---- Layer 3: (16x64) x (64x16[2 live cols]) via WMMA ------------
        v8f acc30 = {}, acc31 = {};
        #pragma unroll
        for (int ks = 0; ks < 2; ++ks) {
            int o0 = col * 64 + ks * 32 + half * 8;
            Frag a0, a1, bW;
            a0.q[0] = *(const uint4*)&hA[o0];
            a0.q[1] = *(const uint4*)&hA[o0 + 16];
            a1.q[0] = *(const uint4*)&hB[o0];
            a1.q[1] = *(const uint4*)&hB[o0 + 16];
            const uint4* bp = (const uint4*)(wof + (((l * 2 + ks) * 32 + lane) << 4));
            bW.q[0] = bp[0]; bW.q[1] = bp[1];
            acc30 = __builtin_amdgcn_wmma_f32_16x16x32_bf16(
                false, a0.v, false, bW.v, (short)0, acc30, false, false);
            acc31 = __builtin_amdgcn_wmma_f32_16x16x32_bf16(
                false, a1.v, false, bW.v, (short)0, acc31, false, false);
        }
        // D column 0 -> mus[:, l+1], column 1 -> lvs[:, l+1]
        if (col < 2) {
            float bb = bout[l * 2 + col];
            int ob = (col == 0) ? 0 : BATCH * 32;
            #pragma unroll
            for (int j = 0; j < 8; ++j) {
                int r0 = rowBase + j + half * 8;
                out[ob + r0 * 32 + (l + 1)]        = acc30[j] + bb;
                out[ob + (r0 + 16) * 32 + (l + 1)] = acc31[j] + bb;
            }
        }
        lds_fence();   // WAR: next iteration overwrites h1/h2 tiles
    }
}

// -------------------------------- launcher ----------------------------------

extern "C" void kernel_launch(void* const* d_in, const int* in_sizes, int n_in,
                              void* d_out, int out_size, void* d_ws, size_t ws_size,
                              hipStream_t stream) {
    const float* z     = (const float*)d_in[0];
    const float* W1    = (const float*)d_in[1];
    const float* b1    = (const float*)d_in[2];
    const float* W2    = (const float*)d_in[3];
    const float* b2    = (const float*)d_in[4];
    const float* Wout  = (const float*)d_in[5];
    const float* bout  = (const float*)d_in[6];
    const float* bout0 = (const float*)d_in[7];
    float* out = (float*)d_out;

    // Workspace: bf16 fragment-ordered weights (~806 KB total).
    unsigned short* w1f = (unsigned short*)d_ws;                       // 253952 B
    unsigned short* w2f = (unsigned short*)((char*)d_ws + 253952);     // 507904 B
    unsigned short* wof = (unsigned short*)((char*)d_ws + 761856);     //  63488 B

    prep_w1<<<496, 256, 0, stream>>>(W1, w1f);     // 31*8*32*16  / 256
    prep_w2<<<992, 256, 0, stream>>>(W2, w2f);     // 31*4*4*32*16 / 256
    prep_wo<<<124, 256, 0, stream>>>(Wout, wof);   // 31*2*32*16  / 256

    // 65536 rows / (32 rows per wave * 4 waves per block) = 512 blocks.
    arprior_main<<<512, 128, 0, stream>>>(z, b1, b2, bout, bout0, w1f, w2f, wof, out);
}
